// QuantizedLinear8bit_74594991997150
// MI455X (gfx1250) — matrix operand order repaired and checked by reading, not compiled, not run on hardware
//
#include <hip/hip_runtime.h>

typedef _Float16 v16h __attribute__((ext_vector_type(16)));
typedef _Float16 v8h  __attribute__((ext_vector_type(8)));
typedef float    v8f  __attribute__((ext_vector_type(8)));
typedef float    v4f  __attribute__((ext_vector_type(4)));
typedef int      v4i  __attribute__((ext_vector_type(4)));

static constexpr int kInF  = 4096;
static constexpr int kOutF = 11008;
static constexpr int kM    = 8192;     // B*S
static constexpr int kBK   = 32;       // K per step (== WMMA K)
static constexpr int kNK   = kInF / kBK;   // 128 (even)
static constexpr int kLDH  = 40;       // LDS row stride in halves (32 + 8 pad, 80B, 16B aligned)
static constexpr int kTileHalves = 128 * kLDH;          // 5120 halves per tile buffer

union Frag16 {
    v16h v;
    struct { v8h lo; v8h hi; } p;
};

// ---------------------------------------------------------------------------
// Shared epilogue: out[m][n] = acc * scale[n] + bias_scale * bq[n]  (NT stores)
// ---------------------------------------------------------------------------
__device__ __forceinline__ void store_epilogue(
    const v8f acc[4][2], int n0, int m0, int wn, int wm, int lmod, int lhalf,
    const float* __restrict__ wscale, const int* __restrict__ bq, float bsc,
    float* __restrict__ out)
{
#pragma unroll
    for (int i = 0; i < 4; ++i) {
        const int nb = n0 + wn * 64 + i * 16 + lhalf * 8;   // 8 consecutive n per lane
        const v4f s0 = *(const v4f*)(wscale + nb);
        const v4f s1 = *(const v4f*)(wscale + nb + 4);
        const v4i b0 = *(const v4i*)(bq + nb);
        const v4i b1 = *(const v4i*)(bq + nb + 4);
#pragma unroll
        for (int j = 0; j < 2; ++j) {
            const int m = m0 + wm * 32 + j * 16 + lmod;
            float* o = out + (size_t)m * kOutF + nb;
            v4f o0, o1;
#pragma unroll
            for (int r = 0; r < 4; ++r) {
                o0[r] = acc[i][j][r]     * s0[r] + bsc * (float)b0[r];
                o1[r] = acc[i][j][4 + r] * s1[r] + bsc * (float)b1[r];
            }
            __builtin_nontemporal_store(o0, (v4f*)o);        // global_store_b128 NT
            __builtin_nontemporal_store(o1, (v4f*)(o + 4));
        }
    }
}

// ---------------------------------------------------------------------------
// Pre-pass 1: dequant weights int32 -> f16 (q - zp), 8 elems/thread
// ---------------------------------------------------------------------------
__global__ __launch_bounds__(256)
void convert_w_kernel(const int* __restrict__ wq, const int* __restrict__ wzp,
                      _Float16* __restrict__ wf)
{
    const size_t e = ((size_t)blockIdx.x * 256 + threadIdx.x) * 8;
    const int zp = wzp[e >> 12];                 // row = e / kInF
    const v4i q0 = __builtin_nontemporal_load((const v4i*)(wq + e));
    const v4i q1 = __builtin_nontemporal_load((const v4i*)(wq + e + 4));
    v8h h;
#pragma unroll
    for (int i = 0; i < 4; ++i) {
        h[i]     = (_Float16)(float)(q0[i] - zp);
        h[4 + i] = (_Float16)(float)(q1[i] - zp);
    }
    *(v8h*)(wf + e) = h;
}

// ---------------------------------------------------------------------------
// Pre-pass 2: activations f32 -> f16, 8 elems/thread
// ---------------------------------------------------------------------------
__global__ __launch_bounds__(256)
void convert_x_kernel(const float* __restrict__ x, _Float16* __restrict__ xf)
{
    const size_t e = ((size_t)blockIdx.x * 256 + threadIdx.x) * 8;
    const v4f x0 = __builtin_nontemporal_load((const v4f*)(x + e));
    const v4f x1 = __builtin_nontemporal_load((const v4f*)(x + e + 4));
    v8h h;
#pragma unroll
    for (int i = 0; i < 4; ++i) {
        h[i]     = (_Float16)x0[i];
        h[4 + i] = (_Float16)x1[i];
    }
    *(v8h*)(xf + e) = h;
}

// ---------------------------------------------------------------------------
// Main GEMM, pure f16 path: async global->LDS copies, no staging VGPRs,
// no conversion VALU in the K-loop.  D(nxm) = Wf16(16x32) * xT(32x16) + C.
// ---------------------------------------------------------------------------
__global__ __launch_bounds__(256)
void qlinear_gemm_f16_async(const _Float16* __restrict__ wf,
                            const _Float16* __restrict__ xf,
                            const float* __restrict__ wscale,
                            const int*   __restrict__ bq,
                            const float* __restrict__ bscale,
                            float*       __restrict__ out)
{
    // Single shared block (offset 0 in LDS alloc): [W buf0][W buf1][X buf0][X buf1]
    __shared__ __align__(16) _Float16 smem[4 * kTileHalves];   // 40 KB

    // CRITICAL: let smem's address escape into opaque asm so the compiler
    // must assume the async-copy asm (with "memory" clobber) writes LDS.
    // Without this, capture analysis proves smem is never stored and the
    // fragment ds_loads get folded to undef (observed in round 2).
    void* smem_escape = (void*)smem;
    asm volatile("" : "+v"(smem_escape) :: "memory");

    const int tid   = threadIdx.x;
    const int lane  = tid & 31;
    const int wave  = tid >> 5;
    const int lmod  = lane & 15;
    const int lhalf = lane >> 4;
    const int wn    = wave & 1;
    const int wm    = wave >> 1;

    const int n0 = blockIdx.x * 128;
    const int m0 = blockIdx.y * 128;

    // staging mapping: 256 threads x 32B cover one 128x32 f16 tile
    const int srow  = tid >> 1;
    const int shalf = tid & 1;
    const unsigned lstB = (unsigned)(srow * kLDH + shalf * 16) * 2;  // bytes, 16B aligned

    const unsigned long long wsrc =
        (unsigned long long)(uintptr_t)(wf + (size_t)(n0 + srow) * kInF + shalf * 16);
    const unsigned long long xsrc =
        (unsigned long long)(uintptr_t)(xf + (size_t)(m0 + srow) * kInF + shalf * 16);

    auto issue_async = [&](int kt, int bufi) {
        // smem is the only __shared__ object -> it sits at LDS offset 0, so
        // these integers are the LDS byte addresses the instruction needs.
        const unsigned wl = (unsigned)(bufi * kTileHalves * 2) + lstB;
        const unsigned xl = (unsigned)((2 + bufi) * kTileHalves * 2) + lstB;
        const unsigned long long wp = wsrc + (unsigned long long)kt * (kBK * 2);
        const unsigned long long xp = xsrc + (unsigned long long)kt * (kBK * 2);
        // INST_OFFSET is added to BOTH the LDS and global address (ISA 08 §4.4),
        // which matches our contiguous 32B-per-thread layout on both sides.
        asm volatile("global_load_async_to_lds_b128 %0, %1, off"
                     :: "v"(wl), "v"(wp) : "memory");
        asm volatile("global_load_async_to_lds_b128 %0, %1, off offset:16"
                     :: "v"(wl), "v"(wp) : "memory");
        asm volatile("global_load_async_to_lds_b128 %0, %1, off"
                     :: "v"(xl), "v"(xp) : "memory");
        asm volatile("global_load_async_to_lds_b128 %0, %1, off offset:16"
                     :: "v"(xl), "v"(xp) : "memory");
    };

    v8f acc[4][2];
#pragma unroll
    for (int i = 0; i < 4; ++i)
#pragma unroll
        for (int j = 0; j < 2; ++j)
            acc[i][j] = (v8f)0.0f;

    auto compute_tile = [&](int bufi) {
        const _Float16* lw = &smem[bufi * kTileHalves];
        const _Float16* lx = &smem[(2 + bufi) * kTileHalves];

        Frag16 a[4];
#pragma unroll
        for (int i = 0; i < 4; ++i) {
            const _Float16* p = lw + (wn * 64 + i * 16 + lmod) * kLDH + lhalf * 8;
            a[i].p.lo = *(const v8h*)p;          // ds_load_b128: K=kb..kb+7
            a[i].p.hi = *(const v8h*)(p + 16);   // ds_load_b128: K=kb+16..kb+23
        }
        Frag16 b[2];
#pragma unroll
        for (int j = 0; j < 2; ++j) {
            const _Float16* p = lx + (wm * 32 + j * 16 + lmod) * kLDH + lhalf * 8;
            b[j].p.lo = *(const v8h*)p;
            b[j].p.hi = *(const v8h*)(p + 16);
        }
#pragma unroll
        for (int i = 0; i < 4; ++i)
#pragma unroll
            for (int j = 0; j < 2; ++j)
                acc[i][j] = __builtin_amdgcn_wmma_f32_16x16x32_f16(
                    false, a[i].v, false, b[j].v, (short)0, acc[i][j], false, false);
    };

    // Prologue: tile 0 into buffer 0.
    issue_async(0, 0);
    asm volatile("s_wait_asynccnt 0" ::: "memory");
    __syncthreads();

    // Explicit ping-pong over buffers 0/1; kNK is even so no tail handling.
    for (int kt = 0; kt < kNK; kt += 2) {
        issue_async(kt + 1, 1);               // next tile -> buffer 1
        compute_tile(0);                      // consume buffer 0
        asm volatile("s_wait_asynccnt 0" ::: "memory");
        __syncthreads();

        if (kt + 2 < kNK) issue_async(kt + 2, 0);   // tile after -> buffer 0
        compute_tile(1);                      // consume buffer 1
        asm volatile("s_wait_asynccnt 0" ::: "memory");
        __syncthreads();
    }

    store_epilogue(acc, n0, m0, wn, wm, lmod, lhalf, wscale, bq, bscale[0], out);
}

// ---------------------------------------------------------------------------
// Fallback: fused dequant GEMM (used when d_ws is too small for f16 copies)
// ---------------------------------------------------------------------------
__global__ __launch_bounds__(256)
void qlinear_gemm_fused(const float* __restrict__ x,
                        const int*   __restrict__ wq,
                        const float* __restrict__ wscale,
                        const int*   __restrict__ wzp,
                        const int*   __restrict__ bq,
                        const float* __restrict__ bscale,
                        float*       __restrict__ out)
{
    __shared__ _Float16 lw[2][kTileHalves];
    __shared__ _Float16 lx[2][kTileHalves];

    const int tid   = threadIdx.x;
    const int lane  = tid & 31;
    const int wave  = tid >> 5;
    const int lmod  = lane & 15;
    const int lhalf = lane >> 4;
    const int wn    = wave & 1;
    const int wm    = wave >> 1;

    const int n0 = blockIdx.x * 128;
    const int m0 = blockIdx.y * 128;

    const int srow  = tid >> 1;
    const int shalf = tid & 1;
    const int zp    = wzp[n0 + srow];

    const int*   wg = wq + (size_t)(n0 + srow) * kInF + shalf * 16;
    const float* xg = x  + (size_t)(m0 + srow) * kInF + shalf * 16;
    const int lst   = srow * kLDH + shalf * 16;

    v4i wr[4];
    v4f xr[4];

    auto g_load = [&](int kk) {
#pragma unroll
        for (int u = 0; u < 4; ++u) {
            wr[u] = *(const v4i*)(wg + kk + u * 4);
            xr[u] = *(const v4f*)(xg + kk + u * 4);
        }
    };
    auto conv_store = [&](int bufi) {
        v8h wo0, wo1, xo0, xo1;
#pragma unroll
        for (int e = 0; e < 8; ++e) {
            wo0[e] = (_Float16)(float)(wr[e >> 2][e & 3] - zp);
            wo1[e] = (_Float16)(float)(wr[2 + (e >> 2)][e & 3] - zp);
            xo0[e] = (_Float16)xr[e >> 2][e & 3];
            xo1[e] = (_Float16)xr[2 + (e >> 2)][e & 3];
        }
        *(v8h*)&lw[bufi][lst]     = wo0;
        *(v8h*)&lw[bufi][lst + 8] = wo1;
        *(v8h*)&lx[bufi][lst]     = xo0;
        *(v8h*)&lx[bufi][lst + 8] = xo1;
    };

    v8f acc[4][2];
#pragma unroll
    for (int i = 0; i < 4; ++i)
#pragma unroll
        for (int j = 0; j < 2; ++j)
            acc[i][j] = (v8f)0.0f;

    g_load(0);
    conv_store(0);

    int buf = 0;
    for (int kt = 0; kt < kNK; ++kt) {
        __syncthreads();
        if (kt + 1 < kNK) g_load((kt + 1) * kBK);
        if (kt + 2 < kNK) {
            __builtin_prefetch(wg + (kt + 2) * kBK, 0, 0);
            __builtin_prefetch(xg + (kt + 2) * kBK, 0, 0);
        }

        Frag16 a[4];
#pragma unroll
        for (int i = 0; i < 4; ++i) {
            const _Float16* p = &lw[buf][(wn * 64 + i * 16 + lmod) * kLDH + lhalf * 8];
            a[i].p.lo = *(const v8h*)p;
            a[i].p.hi = *(const v8h*)(p + 16);
        }
        Frag16 b[2];
#pragma unroll
        for (int j = 0; j < 2; ++j) {
            const _Float16* p = &lx[buf][(wm * 32 + j * 16 + lmod) * kLDH + lhalf * 8];
            b[j].p.lo = *(const v8h*)p;
            b[j].p.hi = *(const v8h*)(p + 16);
        }

#pragma unroll
        for (int i = 0; i < 4; ++i)
#pragma unroll
            for (int j = 0; j < 2; ++j)
                acc[i][j] = __builtin_amdgcn_wmma_f32_16x16x32_f16(
                    false, a[i].v, false, b[j].v, (short)0, acc[i][j], false, false);

        if (kt + 1 < kNK) conv_store(buf ^ 1);
        buf ^= 1;
    }

    store_epilogue(acc, n0, m0, wn, wm, lmod, lhalf, wscale, bq, bscale[0], out);
}

// ---------------------------------------------------------------------------
extern "C" void kernel_launch(void* const* d_in, const int* in_sizes, int n_in,
                              void* d_out, int out_size, void* d_ws, size_t ws_size,
                              hipStream_t stream) {
    const float* x      = (const float*)d_in[0];
    const int*   wq     = (const int*)d_in[1];
    const float* wscale = (const float*)d_in[2];
    const int*   wzp    = (const int*)d_in[3];
    const int*   bq     = (const int*)d_in[4];
    const float* bscale = (const float*)d_in[5];
    float*       out    = (float*)d_out;

    const size_t wElems = (size_t)kOutF * kInF;    // 45,088,768
    const size_t xElems = (size_t)kM * kInF;       // 33,554,432
    const size_t need   = (wElems + xElems) * sizeof(_Float16);  // ~157 MB

    dim3 grid(kOutF / 128, kM / 128);   // 86 x 64 tiles

    if (ws_size >= need) {
        _Float16* wf = (_Float16*)d_ws;
        _Float16* xf = wf + wElems;
        convert_w_kernel<<<(unsigned)(wElems / (256 * 8)), 256, 0, stream>>>(wq, wzp, wf);
        convert_x_kernel<<<(unsigned)(xElems / (256 * 8)), 256, 0, stream>>>(x, xf);
        qlinear_gemm_f16_async<<<grid, 256, 0, stream>>>(wf, xf, wscale, bq, bscale, out);
    } else {
        qlinear_gemm_fused<<<grid, 256, 0, stream>>>(x, wq, wscale, wzp, bq, bscale, out);
    }
}
